// MultiHeadAttention_30253749633731
// MI455X (gfx1250) — matrix operand order repaired and checked by reading, not compiled, not run on hardware
//
#include <hip/hip_runtime.h>
#include <hip/hip_bf16.h>
#include <math.h>

// ---------------------------------------------------------------------------
// MI455X (gfx1250) fused multi-head attention, bf16 WMMA + TDM staging.
// B=4, T=2048, C=384, H=6, D=64.
// ---------------------------------------------------------------------------

#define C_DIM 384
#define H_DIM 6
#define D_DIM 64
#define T_DIM 2048
#define B_DIM 4

typedef __attribute__((ext_vector_type(16))) __bf16 v16bf;
typedef __attribute__((ext_vector_type(8)))  float  v8f;
typedef unsigned int __attribute__((ext_vector_type(4))) u32x4;
typedef int  __attribute__((ext_vector_type(8))) i32x8;
typedef int  __attribute__((ext_vector_type(4))) i32x4;

#if defined(__has_builtin)
#if __has_builtin(__builtin_amdgcn_tensor_load_to_lds)
#define USE_TDM 1
#endif
#endif

#if defined(__has_include)
#if __has_include(<hip/amd_detail/amd_gfx1250_TDM.h>)
#define TDM_6ARG 1   // therock headers -> 6-arg builtin form
#endif
#endif

struct Frag {
    union { v16bf v; uint4 u[2]; };
};

// A-matrix 16x32 bf16 fragment from a row-major tile (ISA 7.12.2):
// half=0 holds K 0..7 / 16..23; half=1 holds K 8..15 / 24..31.
__device__ __forceinline__ Frag load_a(const __bf16* base, int stride, int lane) {
    const int m = lane & 15, half = lane >> 4;
    const __bf16* row = base + m * stride;
    Frag f;
    f.u[0] = *(const uint4*)(row + 8 * half);
    f.u[1] = *(const uint4*)(row + 16 + 8 * half);
    return f;
}

// B-matrix 32x16 bf16 fragment from a K-contiguous ("transposed") tile [n][k]:
// lanes 0-15 hold K 0..7 / 16..23; lanes 16-31 hold K 8..15 / 24..31.
__device__ __forceinline__ Frag load_b(const __bf16* baseT, int stride, int lane) {
    const int n = lane & 15, kh = lane >> 4;
    const __bf16* row = baseT + n * stride;
    Frag f;
    f.u[0] = *(const uint4*)(row + 8 * kh);
    f.u[1] = *(const uint4*)(row + 16 + 8 * kh);
    return f;
}

__device__ __forceinline__ v8f wmma_bf16(const Frag& a, const Frag& b, v8f c) {
    return __builtin_amdgcn_wmma_f32_16x16x32_bf16(false, a.v, false, b.v,
                                                   (short)0, c, false, false);
}

__device__ __forceinline__ unsigned int pack_bf16x2(float a, float b) {
    union { __bf16 h[2]; unsigned int u; } t;
    t.h[0] = (__bf16)a; t.h[1] = (__bf16)b;
    return t.u;
}

__device__ __forceinline__ void wait_tensor0() {
#if defined(__has_builtin) && __has_builtin(__builtin_amdgcn_s_wait_tensorcnt)
    __builtin_amdgcn_s_wait_tensorcnt(0);
#else
    asm volatile("s_wait_tensorcnt 0" ::: "memory");
#endif
}

#if defined(USE_TDM)
// TDM 2-D tile load: tile_h rows of tile_w bf16 elements, source row pitch
// stride_elems, packed contiguously into LDS. D# per CDNA5 ISA §8.3-8.6:
// group0: count=1 | lds_addr | global_addr | type=2; group1: data_size=2B,
// tensor dims == tile dims (no OOB), dim0_stride = pitch. Groups 2/3 zero (2D).
__device__ __forceinline__ void tdm_load_2d(const __bf16* gsrc, __bf16* ldst,
                                            unsigned int tile_w, unsigned int tile_h,
                                            unsigned int stride_elems) {
    unsigned long long ga = (unsigned long long)(uintptr_t)gsrc;
    unsigned int la = (unsigned int)(uintptr_t)ldst;
    u32x4 g0;
    g0.x = 1u;                                               // count=1 (valid)
    g0.y = la;                                               // lds_addr
    g0.z = (unsigned int)ga;                                 // global_addr[31:0]
    g0.w = (unsigned int)((ga >> 32) & 0x01ffffffu) | (2u << 30); // [56:32]|type=2
    i32x8 g1;
    g1[0] = (int)(1u << 16);                                 // data_size=1 -> 2B
    g1[1] = (int)((tile_w & 0xffffu) << 16);                 // tensor_dim0 lo
    g1[2] = (int)((tile_h & 0xffffu) << 16);                 // dim0 hi=0 | dim1 lo
    g1[3] = (int)((tile_w & 0xffffu) << 16);                 // dim1 hi=0 | tile_dim0
    g1[4] = (int)(tile_h & 0xffffu);                         // tile_dim1 | tile_dim2=0
    g1[5] = (int)stride_elems;                               // dim0_stride[31:0]
    g1[6] = 0; g1[7] = 0;
    i32x4 gz = (i32x4){0, 0, 0, 0};
#if defined(TDM_6ARG)
    i32x8 g5 = (i32x8){0, 0, 0, 0, 0, 0, 0, 0};
    __builtin_amdgcn_tensor_load_to_lds(g0, g1, gz, gz, g5, 0);
#else
    __builtin_amdgcn_tensor_load_to_lds(g0, g1, gz, gz, 0);
#endif
}
#endif

// ---------------------------------------------------------------------------
// Kernel 0: one-time weight prep. WT[which][h][d][c] = W[h][c][d] as bf16,
// WoT[d][c] = Wo[c][d] as bf16. Removes all per-chunk gather-transposes.
// ---------------------------------------------------------------------------
__global__ __launch_bounds__(256) void mha_prep_weights(
    const float* __restrict__ Wq, const float* __restrict__ Wk,
    const float* __restrict__ Wv, const float* __restrict__ Wo,
    __bf16* __restrict__ wqT, __bf16* __restrict__ wkT,
    __bf16* __restrict__ wvT, __bf16* __restrict__ woT)
{
    const int NQKV = H_DIM * D_DIM * C_DIM;                  // 147456
    int idx = blockIdx.x * 256 + threadIdx.x;
    if (idx < 3 * NQKV) {
        int which = idx / NQKV, r = idx % NQKV;
        int h = r / (D_DIM * C_DIM), r2 = r % (D_DIM * C_DIM);
        int d = r2 / C_DIM, c = r2 % C_DIM;
        const float* W = which == 0 ? Wq : (which == 1 ? Wk : Wv);
        __bf16* dst = which == 0 ? wqT : (which == 1 ? wkT : wvT);
        dst[r] = (__bf16)W[((size_t)h * C_DIM + c) * D_DIM + d];
    } else {
        int r = idx - 3 * NQKV;
        if (r < C_DIM * C_DIM) {
            int d = r / C_DIM, c = r % C_DIM;
            woT[r] = (__bf16)Wo[(size_t)c * C_DIM + d];
        }
    }
}

// ---------------------------------------------------------------------------
// Kernel 1: q/k/v = x @ W[h]  (fp32 x -> bf16 [B,H,T,64]); weights from WT.
// grid (T/64, H*3, B), block 128 (4 waves).
// ---------------------------------------------------------------------------
__global__ __launch_bounds__(128) void mha_qkv_proj(
    const float* __restrict__ x,
    const __bf16* __restrict__ wqT, const __bf16* __restrict__ wkT,
    const __bf16* __restrict__ wvT,
    __bf16* __restrict__ qb, __bf16* __restrict__ kb, __bf16* __restrict__ vb)
{
    __shared__ __align__(16) __bf16 xLDS[64 * 32];  // [row][c]
    __shared__ __align__(16) __bf16 wT[64 * 32];    // [d][c]

    const int tid = threadIdx.x;
    const int lane = tid & 31, w = tid >> 5;
    const int t0 = blockIdx.x * 64;
    const int h = blockIdx.y / 3, p = blockIdx.y % 3;
    const int b = blockIdx.z;

    const __bf16* WT = (p == 0 ? wqT : (p == 1 ? wkT : wvT)) +
                       (size_t)h * D_DIM * C_DIM;            // [d][c], pitch C
    __bf16* out = (p == 0 ? qb : (p == 1 ? kb : vb)) +
                  ((size_t)(b * H_DIM + h) * T_DIM + t0) * D_DIM;
    const float* xrowbase = x + ((size_t)b * T_DIM + t0) * C_DIM;

    v8f acc[4];
    #pragma unroll
    for (int n = 0; n < 4; ++n) acc[n] = (v8f){0,0,0,0,0,0,0,0};

    for (int kc = 0; kc < C_DIM; kc += 32) {
        // stage W chunk [64][32]: TDM tile copy (already transposed in prep)
#if defined(USE_TDM)
        if (w == 0) {
            tdm_load_2d(WT + kc, wT, 32, 64, C_DIM);
            wait_tensor0();
        }
#else
        #pragma unroll
        for (int j = 0; j < 2; ++j) {
            int i = tid + j * 128;                 // 256 uint4
            int d = i >> 2, q = i & 3;
            *(uint4*)&wT[d * 32 + q * 8] = *(const uint4*)(WT + (size_t)d * C_DIM + kc + q * 8);
        }
#endif
        // stage X tile (64x32) fp32 -> bf16, float4 reads, packed b64 stores
        #pragma unroll
        for (int j = 0; j < 4; ++j) {
            int idx = tid + j * 128;               // 512 float4
            int row = idx >> 3, c4 = idx & 7;
            float4 fv = *(const float4*)(xrowbase + row * C_DIM + kc + c4 * 4);
            uint2 pk;
            pk.x = pack_bf16x2(fv.x, fv.y);
            pk.y = pack_bf16x2(fv.z, fv.w);
            *(uint2*)&xLDS[row * 32 + c4 * 4] = pk;
        }
        __syncthreads();

        Frag a = load_a(&xLDS[(w * 16) * 32], 32, lane);
        #pragma unroll
        for (int n = 0; n < 4; ++n) {
            Frag bb = load_b(&wT[(n * 16) * 32], 32, lane);
            acc[n] = wmma_bf16(a, bb, acc[n]);
        }
        __syncthreads();
    }

    const int half = lane >> 4, nn = lane & 15;
    #pragma unroll
    for (int n = 0; n < 4; ++n)
        #pragma unroll
        for (int r = 0; r < 8; ++r)
            out[(size_t)(w * 16 + r + 8 * half) * D_DIM + n * 16 + nn] = (__bf16)acc[n][r];
}

// ---------------------------------------------------------------------------
// Kernel 2: flash attention per (b,h). grid (T/64, H, B), block 128.
// ---------------------------------------------------------------------------
__global__ __launch_bounds__(128) void mha_flash(
    const __bf16* __restrict__ qb, const __bf16* __restrict__ kb,
    const __bf16* __restrict__ vb, __bf16* __restrict__ ao)
{
    __shared__ __align__(16) __bf16 kLDS[32 * 64];       // [s][d]
    __shared__ __align__(16) __bf16 vT[64 * 32];         // [d][s]
    __shared__ __align__(16) __bf16 pLDS[4 * 16 * 32];   // per-wave P tile

    const int tid = threadIdx.x;
    const int lane = tid & 31, w = tid >> 5;
    const int half = lane >> 4, nn = lane & 15;
    const int h = blockIdx.y, b = blockIdx.z;
    const int t0 = blockIdx.x * 64 + w * 16;
    const float scale = 0.125f;  // 1/sqrt(64)

    const size_t bh = (size_t)(b * H_DIM + h) * T_DIM;
    const __bf16* qg = qb + (bh + t0) * D_DIM;

    Frag qa0 = load_a(qg, D_DIM, lane);
    Frag qa1 = load_a(qg + 32, D_DIM, lane);

    v8f o[4];
    #pragma unroll
    for (int n = 0; n < 4; ++n) o[n] = (v8f){0,0,0,0,0,0,0,0};
    float mrow[8], lrow[8];
    #pragma unroll
    for (int r = 0; r < 8; ++r) { mrow[r] = -3.0e30f; lrow[r] = 0.0f; }

    __bf16* pW = &pLDS[w * 16 * 32];

    for (int sc = 0; sc < T_DIM; sc += 32) {
        // K chunk (32x64 bf16) straight tile: TDM from wave 0
#if defined(USE_TDM)
        if (w == 0) tdm_load_2d(kb + (bh + sc) * D_DIM, kLDS, 64, 32, 64);
#else
        {
            const uint4* ksrc = (const uint4*)(kb + (bh + sc) * D_DIM);
            uint4* kdst = (uint4*)kLDS;
            kdst[tid]       = ksrc[tid];
            kdst[tid + 128] = ksrc[tid + 128];
        }
#endif
        // V chunk transposed manually: vT[d][s] = v[s][d] (vectorized reads)
        {
            const int s = tid & 31, d0 = (tid >> 5) * 16;
            const __bf16* vrow = vb + (bh + sc + s) * D_DIM + d0;
            union { uint4 q; __bf16 hh[8]; } va, vc;
            va.q = *(const uint4*)vrow;
            vc.q = *(const uint4*)(vrow + 8);
            #pragma unroll
            for (int i = 0; i < 8; ++i) {
                vT[(d0 + i) * 32 + s]     = va.hh[i];
                vT[(d0 + 8 + i) * 32 + s] = vc.hh[i];
            }
        }
#if defined(USE_TDM)
        if (w == 0) wait_tensor0();
#endif
        __syncthreads();

        // scores: keys sc..+15 and sc+16..+31, K-dim = D = 64
        Frag kb00 = load_b(&kLDS[0],            D_DIM, lane);
        Frag kb01 = load_b(&kLDS[32],           D_DIM, lane);
        Frag kb10 = load_b(&kLDS[16 * 64],      D_DIM, lane);
        Frag kb11 = load_b(&kLDS[16 * 64 + 32], D_DIM, lane);
        v8f s0 = (v8f){0,0,0,0,0,0,0,0}, s1 = (v8f){0,0,0,0,0,0,0,0};
        s0 = wmma_bf16(qa0, kb00, s0); s0 = wmma_bf16(qa1, kb01, s0);
        s1 = wmma_bf16(qa0, kb10, s1); s1 = wmma_bf16(qa1, kb11, s1);

        // online softmax (rows M = r + 8*half, N over 16-lane groups)
        #pragma unroll
        for (int r = 0; r < 8; ++r) {
            float a0 = s0[r] * scale, a1 = s1[r] * scale;
            float cmax = fmaxf(a0, a1);
            #pragma unroll
            for (int msk = 8; msk >= 1; msk >>= 1)
                cmax = fmaxf(cmax, __shfl_xor(cmax, msk, 32));
            float mnew = fmaxf(mrow[r], cmax);
            float alpha = __expf(mrow[r] - mnew);
            mrow[r] = mnew;
            float p0 = __expf(a0 - mnew), p1 = __expf(a1 - mnew);
            float rs = p0 + p1;
            #pragma unroll
            for (int msk = 8; msk >= 1; msk >>= 1)
                rs += __shfl_xor(rs, msk, 32);
            lrow[r] = lrow[r] * alpha + rs;
            #pragma unroll
            for (int n = 0; n < 4; ++n) o[n][r] *= alpha;
            pW[(r + 8 * half) * 32 + nn]      = (__bf16)p0;
            pW[(r + 8 * half) * 32 + 16 + nn] = (__bf16)p1;
        }
        asm volatile("s_wait_dscnt 0" ::: "memory");  // cross-lane LDS dep in-wave

        Frag pa = load_a(pW, 32, lane);
        #pragma unroll
        for (int n = 0; n < 4; ++n) {
            Frag vfr = load_b(&vT[(n * 16) * 32], 32, lane);
            o[n] = wmma_bf16(pa, vfr, o[n]);
        }
        __syncthreads();
    }

    __bf16* dst = ao + ((size_t)b * T_DIM + t0) * (H_DIM * D_DIM) + h * D_DIM;
    #pragma unroll
    for (int r = 0; r < 8; ++r) {
        float inv = 1.0f / lrow[r];
        #pragma unroll
        for (int n = 0; n < 4; ++n)
            dst[(size_t)(r + 8 * half) * (H_DIM * D_DIM) + n * 16 + nn] =
                (__bf16)(o[n][r] * inv);
    }
}

// ---------------------------------------------------------------------------
// Kernel 3: out = attn_out @ Wo + bo  (A bf16, WoT bf16 pre-transposed).
// grid (B*T/64, 384/96), block 128; wave: 16 rows x 6 n-tiles.
// ---------------------------------------------------------------------------
__global__ __launch_bounds__(128) void mha_out_proj(
    const __bf16* __restrict__ ao, const __bf16* __restrict__ woTW,
    const float* __restrict__ bo, float* __restrict__ out)
{
    __shared__ __align__(16) __bf16 aLDS[64 * 32];  // [row][c]
    __shared__ __align__(16) __bf16 woT[96 * 32];   // [d][c]

    const int tid = threadIdx.x;
    const int lane = tid & 31, w = tid >> 5;
    const int half = lane >> 4, nn = lane & 15;
    const int t0 = blockIdx.x * 64;
    const int g0 = blockIdx.y * 96;

    v8f acc[6];
    #pragma unroll
    for (int n = 0; n < 6; ++n) acc[n] = (v8f){0,0,0,0,0,0,0,0};

    for (int kc = 0; kc < C_DIM; kc += 32) {
#if defined(USE_TDM)
        if (w == 0) {
            tdm_load_2d(ao + (size_t)t0 * C_DIM + kc,   aLDS, 32, 64, C_DIM);
            tdm_load_2d(woTW + (size_t)g0 * C_DIM + kc, woT,  32, 96, C_DIM);
            wait_tensor0();
        }
#else
        #pragma unroll
        for (int j = 0; j < 2; ++j) {
            int idx = tid + j * 128;               // 256 uint4
            int row = idx >> 2, q = idx & 3;
            *(uint4*)&aLDS[row * 32 + q * 8] =
                *(const uint4*)(ao + (size_t)(t0 + row) * C_DIM + kc + q * 8);
        }
        #pragma unroll
        for (int j = 0; j < 3; ++j) {
            int idx = tid + j * 128;               // 384 uint4
            int row = idx >> 2, q = idx & 3;
            *(uint4*)&woT[row * 32 + q * 8] =
                *(const uint4*)(woTW + (size_t)(g0 + row) * C_DIM + kc + q * 8);
        }
#endif
        __syncthreads();

        Frag a = load_a(&aLDS[(w * 16) * 32], 32, lane);
        #pragma unroll
        for (int n = 0; n < 6; ++n) {
            Frag bb = load_b(&woT[(n * 16) * 32], 32, lane);
            acc[n] = wmma_bf16(a, bb, acc[n]);
        }
        __syncthreads();
    }

    #pragma unroll
    for (int n = 0; n < 6; ++n) {
        int col = g0 + n * 16 + nn;
        float bias = bo[col];
        #pragma unroll
        for (int r = 0; r < 8; ++r) {
            int rowg = t0 + w * 16 + r + 8 * half;
            out[(size_t)rowg * C_DIM + col] = acc[n][r] + bias;
        }
    }
}

// ---------------------------------------------------------------------------
extern "C" void kernel_launch(void* const* d_in, const int* in_sizes, int n_in,
                              void* d_out, int out_size, void* d_ws, size_t ws_size,
                              hipStream_t stream) {
    const float* x  = (const float*)d_in[0];
    const float* Wq = (const float*)d_in[1];
    const float* Wk = (const float*)d_in[2];
    const float* Wv = (const float*)d_in[3];
    const float* Wo = (const float*)d_in[4];
    const float* bo = (const float*)d_in[5];
    float* out = (float*)d_out;

    const size_t nqkv = (size_t)B_DIM * H_DIM * T_DIM * D_DIM;   // per tensor
    const size_t nw   = (size_t)H_DIM * D_DIM * C_DIM;           // per proj weight
    __bf16* qb  = (__bf16*)d_ws;
    __bf16* kbf = qb + nqkv;
    __bf16* vbf = kbf + nqkv;
    __bf16* ao  = vbf + nqkv;
    __bf16* wqT = ao + (size_t)B_DIM * T_DIM * C_DIM;
    __bf16* wkT = wqT + nw;
    __bf16* wvT = wkT + nw;
    __bf16* woT = wvT + nw;
    (void)in_sizes; (void)n_in; (void)out_size; (void)ws_size;

    dim3 blk(128);
    {   // weight prep: 3*H*D*C + C*C = 589824 elements
        int total = 3 * (int)nw + C_DIM * C_DIM;
        hipLaunchKernelGGL(mha_prep_weights, dim3((total + 255) / 256), dim3(256),
                           0, stream, Wq, Wk, Wv, Wo, wqT, wkT, wvT, woT);
    }

    dim3 g1(T_DIM / 64, H_DIM * 3, B_DIM);
    hipLaunchKernelGGL(mha_qkv_proj, g1, blk, 0, stream, x, wqT, wkT, wvT, qb, kbf, vbf);

    dim3 g2(T_DIM / 64, H_DIM, B_DIM);
    hipLaunchKernelGGL(mha_flash, g2, blk, 0, stream, qb, kbf, vbf, ao);

    dim3 g3(B_DIM * T_DIM / 64, C_DIM / 96);
    hipLaunchKernelGGL(mha_out_proj, g3, blk, 0, stream, ao, woT, bo, out);
}
